// PainnDensityModel_65695819760302
// MI455X (gfx1250) — matrix operand order repaired and checked by reading, not compiled, not run on hardware
//
#include <hip/hip_runtime.h>
#include <hip/hip_bf16.h>
#include <math.h>

#define Nn     16384          // B*NA nodes
#define Hh     128
#define DEG    20
#define Ee     (Nn*DEG)
#define Kk     20
#define Ll     3
#define Tt     3
#define HEADS  4
#define DHd    32
#define INNER  128
#define NAa    256
#define PI_F   3.14159265358979323846f

typedef __attribute__((ext_vector_type(16))) _Float16      v16h;
typedef __attribute__((ext_vector_type(8)))  float         v8f;
typedef __attribute__((ext_vector_type(4)))  unsigned int  v4u;
typedef __attribute__((ext_vector_type(8)))  int           v8i;
typedef __attribute__((ext_vector_type(4)))  int           v4i;

#define CVT4(f, b, a) { (f)[(b)] = (_Float16)(a).x; (f)[(b)+1] = (_Float16)(a).y; \
                        (f)[(b)+2] = (_Float16)(a).z; (f)[(b)+3] = (_Float16)(a).w; }

union Frag16 { uint4 u[2]; v16h v; };

// ---------------------------------------------------------------------------
// weight prep: W[K x Nc] f32  ->  Wt[Nc x K] f16 (transposed, B-fragment friendly)
// ---------------------------------------------------------------------------
__global__ void wcvt_kernel(const float* __restrict__ W, _Float16* __restrict__ Wt,
                            int K, int Nc)
{
  int i = blockIdx.x * blockDim.x + threadIdx.x;
  if (i >= K * Nc) return;
  int k = i / Nc, n = i % Nc;
  Wt[(size_t)n * K + k] = (_Float16)W[i];
}

// ---------------------------------------------------------------------------
// WMMA GEMM:  C[M x Nc] = act(A[M x K] @ W + bias), W given transposed f16.
// block = 256 threads (8 waves); wave owns 16 rows; grid.x = M/128.
// K compile-time (128/256) so all A fragments stay in registers.
// ACT: 0 none, 1 silu, 2 exact gelu
// ---------------------------------------------------------------------------
template<int ACT, int K>
__global__ __launch_bounds__(256) void gemm_wmma_kernel(
    const float* __restrict__ A, const _Float16* __restrict__ Wt,
    const float* __restrict__ bias, float* __restrict__ C,
    int M, int Nc)
{
  const int wave = threadIdx.x >> 5;
  const int lane = threadIdx.x & 31;
  const int mBase = blockIdx.x * 128 + wave * 16;
  if (mBase >= M) return;                 // wave-uniform
  const int mRow = lane & 15;
  const int half = lane >> 4;

  // ---- load all A fragments for this 16-row strip (K/32 of them)
  const float* Abase = A + (size_t)(mBase + mRow) * K;
  v16h af[K / 32];
  #pragma unroll
  for (int kk = 0; kk < K / 32; ++kk) {
    const float* Arow = Abase + kk * 32 + (half << 3);
    float4 a0 = *(const float4*)(Arow);
    float4 a1 = *(const float4*)(Arow + 4);
    float4 a2 = *(const float4*)(Arow + 16);
    float4 a3 = *(const float4*)(Arow + 20);
    v16h t;
    CVT4(t, 0, a0); CVT4(t, 4, a1); CVT4(t, 8, a2); CVT4(t, 12, a3);
    af[kk] = t;
  }

  for (int nt = 0; nt < Nc; nt += 16) {
    const int col = nt + (lane & 15);
    v8f acc = {};
    #pragma unroll
    for (int kk = 0; kk < K / 32; ++kk) {
      // B fragment: lane=col, k = 8*half+{0..7}, 16+8*half+{0..7}: two 16B loads from transposed f16 W
      const _Float16* Wp = Wt + (size_t)col * K + kk * 32 + (half << 3);
      Frag16 B_;
      B_.u[0] = *(const uint4*)(Wp);
      B_.u[1] = *(const uint4*)(Wp + 16);
      acc = __builtin_amdgcn_wmma_f32_16x16x32_f16(
          false, af[kk], false, B_.v, (short)0, acc, false, false);
    }
    const float bv = bias ? bias[col] : 0.0f;
    #pragma unroll
    for (int r = 0; r < 8; ++r) {
      float val = acc[r] + bv;
      if (ACT == 1)      val = val / (1.0f + __expf(-val));              // silu
      else if (ACT == 2) val = 0.5f * val * (1.0f + erff(val * 0.7071067811865476f));
      C[(size_t)(mBase + r + 8 * half) * Nc + col] = val;
    }
  }
}

// ---------------------------------------------------------------------------
// Node embedding + zero v
// ---------------------------------------------------------------------------
__global__ void embed_kernel(const float* __restrict__ emb, const int* __restrict__ z,
                             float* __restrict__ s, float* __restrict__ vA)
{
  int i = blockIdx.x * blockDim.x + threadIdx.x;   // N*128
  int n = i >> 7, c = i & 127;
  s[i] = emb[(size_t)z[n] * Hh + c];
  size_t b = (size_t)n * 384;
  vA[b + c] = 0.f; vA[b + 128 + c] = 0.f; vA[b + 256 + c] = 0.f;
}

// ---------------------------------------------------------------------------
// Edge geometry: d, unit(3), rbf(20), fc
// ---------------------------------------------------------------------------
__global__ void edge_geom_kernel(const float* __restrict__ xyz, const int* __restrict__ edges,
                                 float* __restrict__ unit, float* __restrict__ rbf,
                                 float* __restrict__ fcb)
{
  int e = blockIdx.x * blockDim.x + threadIdx.x;
  if (e >= Ee) return;
  int sn = edges[2 * e], rn = edges[2 * e + 1];
  float dx = xyz[3 * sn]     - xyz[3 * rn];
  float dy = xyz[3 * sn + 1] - xyz[3 * rn + 1];
  float dz = xyz[3 * sn + 2] - xyz[3 * rn + 2];
  float d = sqrtf(dx * dx + dy * dy + dz * dz + 1e-12f);
  float inv = 1.0f / d;
  unit[(size_t)e * 3]     = dx * inv;
  unit[(size_t)e * 3 + 1] = dy * inv;
  unit[(size_t)e * 3 + 2] = dz * inv;
  const float w = PI_F / 5.0f;
  #pragma unroll
  for (int k = 1; k <= Kk; ++k)
    rbf[(size_t)e * Kk + (k - 1)] = sinf((float)k * w * d) * inv;
  fcb[e] = (d < 5.0f) ? 0.5f * (cosf(w * d) + 1.0f) : 0.0f;
}

// ---------------------------------------------------------------------------
// Fused message pass. 128 threads (thread = channel), 8 nodes per block.
// Wf (20x384 f32, 30KB) staged in LDS via the Tensor Data Mover.
// ---------------------------------------------------------------------------
#define NPB 8
__global__ __launch_bounds__(128) void message_kernel(
    const float* __restrict__ phi, const float* __restrict__ vIn,
    const int* __restrict__ edges, const float* __restrict__ unit,
    const float* __restrict__ rbf, const float* __restrict__ fcb,
    const float* __restrict__ Wf,  const float* __restrict__ bf,
    float* __restrict__ s, float* __restrict__ vOut)
{
  __shared__ float Wfs[Kk * 384];
  __shared__ float bfs[384];
  __shared__ float rbfs[Kk];
  __shared__ float un[3];
  __shared__ float fcs;
  __shared__ int   sendS;

  const int c = threadIdx.x;                 // 0..127

#if __has_builtin(__builtin_amdgcn_tensor_load_to_lds) && __has_builtin(__builtin_amdgcn_s_wait_tensorcnt)
  if (c < 32) {
    // Tensor DMA descriptor (ISA 8.3/8.4): 2-D tensor 384x20 f32, full tile.
    unsigned ldsoff = (unsigned)(uintptr_t)(&Wfs[0]);
    unsigned long long ga = (unsigned long long)(uintptr_t)Wf;
    v4u g0;
    g0.x = 1u;                                        // count=1 (valid, user D#)
    g0.y = ldsoff;                                    // lds_addr
    g0.z = (unsigned)(ga & 0xffffffffull);            // global_addr[31:0]
    g0.w = (unsigned)((ga >> 32) & 0x1ffffffull) | (2u << 30);  // addr[56:32] | type=2
    v8i g1;
    g1[0] = (int)(2u << 16);                          // workgroup_mask=0, data_size=4B
    g1[1] = (int)((384u & 0xffffu) << 16);            // tensor_dim0 lo16 (=384)
    g1[2] = (int)(((384u >> 16) & 0xffffu) | ((20u & 0xffffu) << 16)); // dim0 hi | dim1 lo (=20)
    g1[3] = (int)(((20u >> 16) & 0xffffu) | (384u << 16));             // dim1 hi | tile_dim0=384
    g1[4] = (int)(20u);                               // tile_dim1=20, tile_dim2=0
    g1[5] = (int)(384u);                              // tensor_dim0_stride lo32 = 384
    g1[6] = 0;                                        // stride hi / dim1_stride lo
    g1[7] = 0;
    v4i zz = {0, 0, 0, 0};
    v8i z8 = {0, 0, 0, 0, 0, 0, 0, 0};
    __builtin_amdgcn_tensor_load_to_lds(g0, g1, zz, zz, z8, 0);
    __builtin_amdgcn_s_wait_tensorcnt(0);
  }
#else
  for (int i = c; i < Kk * 384; i += 128) Wfs[i] = Wf[i];
#endif
  for (int i = c; i < 384; i += 128) bfs[i] = bf[i];
  __syncthreads();

  const int n0 = blockIdx.x * NPB;
  for (int n = n0; n < n0 + NPB; ++n) {
    float sAcc = 0.f, vAcc0 = 0.f, vAcc1 = 0.f, vAcc2 = 0.f;
    for (int j = 0; j < DEG; ++j) {
      const int e = n * DEG + j;
      __syncthreads();
      if (c < Kk)                 rbfs[c]     = rbf[(size_t)e * Kk + c];
      if (c >= Kk && c < Kk + 3)  un[c - Kk]  = unit[(size_t)e * 3 + (c - Kk)];
      if (c == 24)                fcs         = fcb[e];
      if (c == 25)                sendS       = edges[(size_t)e * 2];
      __syncthreads();
      const int send = sendS;
      float f0 = bfs[c], f1 = bfs[128 + c], f2 = bfs[256 + c];
      #pragma unroll
      for (int t = 0; t < Kk; ++t) {
        const float r = rbfs[t];
        f0 += r * Wfs[t * 384 + c];
        f1 += r * Wfs[t * 384 + 128 + c];
        f2 += r * Wfs[t * 384 + 256 + c];
      }
      const float fcv = fcs;
      f0 *= fcv; f1 *= fcv; f2 *= fcv;
      const float* ps = phi + (size_t)send * 384;
      const float dms  = ps[c]       * f0;
      const float dmv1 = ps[128 + c] * f1;
      const float dmv2 = ps[256 + c] * f2;
      const float* vs = vIn + (size_t)send * 384;
      sAcc  += dms;
      vAcc0 += vs[c]       * dmv1 + un[0] * dmv2;
      vAcc1 += vs[128 + c] * dmv1 + un[1] * dmv2;
      vAcc2 += vs[256 + c] * dmv1 + un[2] * dmv2;
    }
    s[(size_t)n * Hh + c] += sAcc;
    const float* vi = vIn  + (size_t)n * 384;
    float*       vo = vOut + (size_t)n * 384;
    vo[c]       = vi[c]       + vAcc0;
    vo[128 + c] = vi[128 + c] + vAcc1;
    vo[256 + c] = vi[256 + c] + vAcc2;
  }
}

// ---------------------------------------------------------------------------
// pack: Vn = sqrt(sum_c V^2 + eps); dot = sum_c U*V; cat = [s, Vn]
// ---------------------------------------------------------------------------
__global__ void pack_kernel(const float* __restrict__ s, const float* __restrict__ U,
                            const float* __restrict__ V, float* __restrict__ cat,
                            float* __restrict__ dot)
{
  int i = blockIdx.x * blockDim.x + threadIdx.x;   // N*128
  int n = i >> 7, j = i & 127;
  size_t b = (size_t)n * 384;
  float u0 = U[b + j], u1 = U[b + 128 + j], u2 = U[b + 256 + j];
  float v0 = V[b + j], v1 = V[b + 128 + j], v2 = V[b + 256 + j];
  dot[i] = u0 * v0 + u1 * v1 + u2 * v2;
  cat[(size_t)n * 256 + j]       = s[i];
  cat[(size_t)n * 256 + 128 + j] = sqrtf(v0 * v0 + v1 * v1 + v2 * v2 + 1e-12f);
}

// combine: s += ass + asv*dot ; vA = vB + U*avv
__global__ void combine_kernel(const float* __restrict__ a, const float* __restrict__ dot,
                               const float* __restrict__ U, const float* __restrict__ vB,
                               float* __restrict__ s, float* __restrict__ vA)
{
  int i = blockIdx.x * blockDim.x + threadIdx.x;   // N*128
  int n = i >> 7, j = i & 127;
  size_t b = (size_t)n * 384;
  float ass = a[b + j], asv = a[b + 128 + j], avv = a[b + 256 + j];
  s[i] += ass + asv * dot[i];
  #pragma unroll
  for (int c = 0; c < 3; ++c) {
    size_t idx = b + (size_t)c * 128 + j;
    vA[idx] = vB[idx] + U[idx] * avv;
  }
}

// ---------------------------------------------------------------------------
// fused triple LayerNorm: xl = LN(x; ln), h = LN(xl; a_ln), fx = LN(xl; f_ln)
// ---------------------------------------------------------------------------
__global__ __launch_bounds__(256) void ln3_kernel(
    const float* __restrict__ x,
    const float* __restrict__ lnw, const float* __restrict__ lnb,
    const float* __restrict__ aw,  const float* __restrict__ ab_,
    const float* __restrict__ fw,  const float* __restrict__ fb,
    float* __restrict__ xl, float* __restrict__ h, float* __restrict__ fx)
{
  const int lane = threadIdx.x & 31;
  const int n = blockIdx.x * 8 + (threadIdx.x >> 5);
  const float* xr = x + (size_t)n * Hh;
  float v[4]; float m = 0.f;
  #pragma unroll
  for (int i = 0; i < 4; ++i) { v[i] = xr[lane + 32 * i]; m += v[i]; }
  #pragma unroll
  for (int o = 16; o; o >>= 1) m += __shfl_xor(m, o, 32);
  m *= (1.f / 128.f);
  float var = 0.f;
  #pragma unroll
  for (int i = 0; i < 4; ++i) { float d = v[i] - m; var += d * d; }
  #pragma unroll
  for (int o = 16; o; o >>= 1) var += __shfl_xor(var, o, 32);
  float inv = rsqrtf(var * (1.f / 128.f) + 1e-5f);
  float xv[4]; float m2 = 0.f;
  #pragma unroll
  for (int i = 0; i < 4; ++i) {
    int c = lane + 32 * i;
    xv[i] = (v[i] - m) * inv * lnw[c] + lnb[c];
    xl[(size_t)n * Hh + c] = xv[i];
    m2 += xv[i];
  }
  #pragma unroll
  for (int o = 16; o; o >>= 1) m2 += __shfl_xor(m2, o, 32);
  m2 *= (1.f / 128.f);
  float var2 = 0.f;
  #pragma unroll
  for (int i = 0; i < 4; ++i) { float d = xv[i] - m2; var2 += d * d; }
  #pragma unroll
  for (int o = 16; o; o >>= 1) var2 += __shfl_xor(var2, o, 32);
  float inv2 = rsqrtf(var2 * (1.f / 128.f) + 1e-5f);
  #pragma unroll
  for (int i = 0; i < 4; ++i) {
    int c = lane + 32 * i;
    float nn = (xv[i] - m2) * inv2;
    h [(size_t)n * Hh + c] = nn * aw[c] + ab_[c];
    fx[(size_t)n * Hh + c] = nn * fw[c] + fb[c];
  }
}

// single LayerNorm (final)
__global__ __launch_bounds__(256) void ln1_kernel(
    const float* __restrict__ x, const float* __restrict__ lnw,
    const float* __restrict__ lnb, float* __restrict__ y)
{
  const int lane = threadIdx.x & 31;
  const int n = blockIdx.x * 8 + (threadIdx.x >> 5);
  const float* xr = x + (size_t)n * Hh;
  float v[4]; float m = 0.f;
  #pragma unroll
  for (int i = 0; i < 4; ++i) { v[i] = xr[lane + 32 * i]; m += v[i]; }
  #pragma unroll
  for (int o = 16; o; o >>= 1) m += __shfl_xor(m, o, 32);
  m *= (1.f / 128.f);
  float var = 0.f;
  #pragma unroll
  for (int i = 0; i < 4; ++i) { float d = v[i] - m; var += d * d; }
  #pragma unroll
  for (int o = 16; o; o >>= 1) var += __shfl_xor(var, o, 32);
  float inv = rsqrtf(var * (1.f / 128.f) + 1e-5f);
  #pragma unroll
  for (int i = 0; i < 4; ++i) {
    int c = lane + 32 * i;
    y[(size_t)n * Hh + c] = (v[i] - m) * inv * lnw[c] + lnb[c];
  }
}

// ---------------------------------------------------------------------------
// per-head k/v LayerNorm over DH=32; writes f16 q (plain), k (LN) row-major
// per-head and v (LN) transposed [bh][d][key] for WMMA attention.
// one wave per (node, head)
// ---------------------------------------------------------------------------
__global__ __launch_bounds__(256) void knvn_kernel(
    const float* __restrict__ qkv,
    const float* __restrict__ kw, const float* __restrict__ kb,
    const float* __restrict__ vw, const float* __restrict__ vb,
    _Float16* __restrict__ q16, _Float16* __restrict__ k16,
    _Float16* __restrict__ v16t)
{
  const int lane = threadIdx.x & 31;
  const int row  = blockIdx.x * 8 + (threadIdx.x >> 5);  // n*HEADS+h
  const int n = row >> 2, h = row & 3;
  const size_t base = (size_t)n * 384 + h * DHd + lane;
  float qq = qkv[base];
  float kq = qkv[base + INNER];
  float vq = qkv[base + 2 * INNER];
  float mk = kq, mv = vq;
  #pragma unroll
  for (int o = 16; o; o >>= 1) { mk += __shfl_xor(mk, o, 32); mv += __shfl_xor(mv, o, 32); }
  mk *= (1.f / 32.f); mv *= (1.f / 32.f);
  float dk = kq - mk, dv = vq - mv;
  float vk = dk * dk, vv = dv * dv;
  #pragma unroll
  for (int o = 16; o; o >>= 1) { vk += __shfl_xor(vk, o, 32); vv += __shfl_xor(vv, o, 32); }
  float ik = rsqrtf(vk * (1.f / 32.f) + 1e-5f);
  float iv = rsqrtf(vv * (1.f / 32.f) + 1e-5f);
  q16[(size_t)n * INNER + h * DHd + lane] = (_Float16)qq;
  k16[(size_t)n * INNER + h * DHd + lane] = (_Float16)(dk * ik * kw[lane] + kb[lane]);
  const int bh = (n >> 8) * HEADS + h;     // batch*HEADS + h
  v16t[((size_t)bh * DHd + lane) * NAa + (n & 255)] = (_Float16)(dv * iv * vw[lane] + vb[lane]);
}

// ---------------------------------------------------------------------------
// WMMA attention: one block (4 waves) per (batch, head).
// wave handles 4 strips of 16 query rows; scores 16x256 via 16 WMMAs (K=32),
// row softmax in C-layout registers, P (f16) spilled to wave-private LDS,
// P@V via WMMA against transposed f16 V.
// ---------------------------------------------------------------------------
__global__ __launch_bounds__(128) void attn_wmma_kernel(
    const _Float16* __restrict__ q16, const _Float16* __restrict__ k16,
    const _Float16* __restrict__ v16t, float* __restrict__ o)
{
  __shared__ __align__(16) _Float16 Pl[4 * 16 * 256];   // 32 KB
  const int bh = blockIdx.x;                  // b*HEADS + h
  const int b = bh >> 2, h = bh & 3;
  const int nb = b * NAa;
  const int wave = threadIdx.x >> 5, lane = threadIdx.x & 31;
  const int lane15 = lane & 15, half = lane >> 4;
  _Float16* pl = Pl + wave * 16 * 256;

  for (int st = 0; st < 4; ++st) {
    const int m0 = st * 64 + wave * 16;
    // ---- q A-fragment (16 rows x K=32)
    const _Float16* qr = q16 + (size_t)(nb + m0 + lane15) * INNER + h * DHd;
    Frag16 Aq;
    Aq.u[0] = *(const uint4*)(qr + 8 * half);
    Aq.u[1] = *(const uint4*)(qr + 16 + 8 * half);

    // ---- scores: 16 key tiles
    v8f acc[16];
    #pragma unroll
    for (int jt = 0; jt < 16; ++jt) {
      const _Float16* kr = k16 + (size_t)(nb + jt * 16 + lane15) * INNER + h * DHd + 8 * half;
      Frag16 Bk;
      Bk.u[0] = *(const uint4*)(kr);
      Bk.u[1] = *(const uint4*)(kr + 16);
      v8f z = {};
      acc[jt] = __builtin_amdgcn_wmma_f32_16x16x32_f16(
          false, Aq.v, false, Bk.v, (short)0, z, false, false);
    }
    // ---- row-wise softmax (row = r + 8*half, cols = lane15 across 16 tiles)
    #pragma unroll
    for (int r = 0; r < 8; ++r) {
      float mx = -1e30f;
      #pragma unroll
      for (int jt = 0; jt < 16; ++jt) mx = fmaxf(mx, acc[jt][r]);
      #pragma unroll
      for (int off = 8; off; off >>= 1) mx = fmaxf(mx, __shfl_xor(mx, off, 32));
      float su = 0.f;
      #pragma unroll
      for (int jt = 0; jt < 16; ++jt) {
        float e = __expf((acc[jt][r] - mx) * 0.17677669529663687f);   // 1/sqrt(32)
        acc[jt][r] = e; su += e;
      }
      #pragma unroll
      for (int off = 8; off; off >>= 1) su += __shfl_xor(su, off, 32);
      const float inv = 1.f / su;
      #pragma unroll
      for (int jt = 0; jt < 16; ++jt)
        pl[(r + 8 * half) * 256 + jt * 16 + lane15] = (_Float16)(acc[jt][r] * inv);
    }
    // ---- P @ V : two 16-wide d tiles, K=256
    #pragma unroll
    for (int dt = 0; dt < 2; ++dt) {
      v8f oa = {};
      #pragma unroll
      for (int k0 = 0; k0 < 256; k0 += 32) {
        const _Float16* pr = pl + lane15 * 256 + k0;
        Frag16 Ap;
        Ap.u[0] = *(const uint4*)(pr + 8 * half);
        Ap.u[1] = *(const uint4*)(pr + 16 + 8 * half);
        const _Float16* vr = v16t + ((size_t)bh * DHd + dt * 16 + lane15) * NAa + k0 + 8 * half;
        Frag16 Bv;
        Bv.u[0] = *(const uint4*)(vr);
        Bv.u[1] = *(const uint4*)(vr + 16);
        oa = __builtin_amdgcn_wmma_f32_16x16x32_f16(
            false, Ap.v, false, Bv.v, (short)0, oa, false, false);
      }
      #pragma unroll
      for (int r = 0; r < 8; ++r)
        o[(size_t)(nb + m0 + r + 8 * half) * INNER + h * DHd + dt * 16 + lane15] = oa[r];
    }
  }
}

// residual: x = xl + o2 + f2
__global__ void residual_kernel(const float* __restrict__ xl, const float* __restrict__ o2,
                                const float* __restrict__ f2, float* __restrict__ x)
{
  int i = blockIdx.x * blockDim.x + threadIdx.x;
  x[i] = xl[i] + o2[i] + f2[i];
}

// readout: out[n] = dot(r1[n,:], w2) + b2  (wave per node)
__global__ __launch_bounds__(256) void rowdot_kernel(
    const float* __restrict__ r1, const float* __restrict__ w2,
    const float* __restrict__ b2, float* __restrict__ out)
{
  const int lane = threadIdx.x & 31;
  const int n = blockIdx.x * 8 + (threadIdx.x >> 5);
  const float* r = r1 + (size_t)n * Hh;
  float acc = 0.f;
  #pragma unroll
  for (int i = 0; i < 4; ++i) acc += r[lane + 32 * i] * w2[lane + 32 * i];
  #pragma unroll
  for (int o = 16; o; o >>= 1) acc += __shfl_xor(acc, o, 32);
  if (lane == 0) out[n] = acc + b2[0];
}

// ---------------------------------------------------------------------------
extern "C" void kernel_launch(void* const* d_in, const int* in_sizes, int n_in,
                              void* d_out, int out_size, void* d_ws, size_t ws_size,
                              hipStream_t stream)
{
  (void)in_sizes; (void)n_in; (void)out_size; (void)ws_size;
  const float* xyz    = (const float*)d_in[0];
  const int*   node_z = (const int*)  d_in[1];
  const int*   edges  = (const int*)  d_in[2];
  const float* emb    = (const float*)d_in[3];
  const float* iW1    = (const float*)d_in[4];
  const float* ib1    = (const float*)d_in[5];
  const float* iW2    = (const float*)d_in[6];
  const float* ib2    = (const float*)d_in[7];
  const float* iWf    = (const float*)d_in[8];
  const float* ibf    = (const float*)d_in[9];
  const float* uWu    = (const float*)d_in[10];
  const float* uWv    = (const float*)d_in[11];
  const float* uWm1   = (const float*)d_in[12];
  const float* ubm1   = (const float*)d_in[13];
  const float* uWm2   = (const float*)d_in[14];
  const float* ubm2   = (const float*)d_in[15];
  const float* ln_w   = (const float*)d_in[16];
  const float* ln_b   = (const float*)d_in[17];
  const float* a_ln_w = (const float*)d_in[18];
  const float* a_ln_b = (const float*)d_in[19];
  const float* Wqkv   = (const float*)d_in[20];
  const float* k_ln_w = (const float*)d_in[21];
  const float* k_ln_b = (const float*)d_in[22];
  const float* v_ln_w = (const float*)d_in[23];
  const float* v_ln_b = (const float*)d_in[24];
  const float* Wo     = (const float*)d_in[25];
  const float* bo     = (const float*)d_in[26];
  const float* f_ln_w = (const float*)d_in[27];
  const float* f_ln_b = (const float*)d_in[28];
  const float* fW1    = (const float*)d_in[29];
  const float* fb1    = (const float*)d_in[30];
  const float* fW2    = (const float*)d_in[31];
  const float* fb2    = (const float*)d_in[32];
  const float* rW1    = (const float*)d_in[33];
  const float* rb1    = (const float*)d_in[34];
  const float* rW2    = (const float*)d_in[35];
  const float* rb2    = (const float*)d_in[36];

  size_t off = 0;
  auto allocB = [&](size_t bytes) {
    void* p = (char*)d_ws + off;
    off += (bytes + 255) & ~(size_t)255;
    return p;
  };
  float* s    = (float*)allocB((size_t)Nn * Hh  * 4);  // node scalars / transformer x
  float* vA   = (float*)allocB((size_t)Nn * 384 * 4);  // vector feats (ping)
  float* vB   = (float*)allocB((size_t)Nn * 384 * 4);  // vector feats (pong)
  float* U    = (float*)allocB((size_t)Nn * 384 * 4);
  float* V    = (float*)allocB((size_t)Nn * 384 * 4);
  float* unit = (float*)allocB((size_t)Ee * 3  * 4);
  float* rbf  = (float*)allocB((size_t)Ee * Kk * 4);
  float* fcb  = (float*)allocB((size_t)Ee * 4);
  float* b384 = (float*)allocB((size_t)Nn * 384 * 4);  // phi / a / qkv
  float* cat  = (float*)allocB((size_t)Nn * 256 * 4);  // [s,Vn]
  float* bufA = (float*)allocB((size_t)Nn * Hh * 4);   // A1 / g / r1
  float* bufB = (float*)allocB((size_t)Nn * Hh * 4);   // dot / f2
  float* xl   = (float*)allocB((size_t)Nn * Hh * 4);
  float* hb   = (float*)allocB((size_t)Nn * Hh * 4);   // h / o
  float* fxb  = (float*)allocB((size_t)Nn * Hh * 4);   // fx / o2
  _Float16* q16  = (_Float16*)allocB((size_t)Nn * INNER * 2);
  _Float16* k16  = (_Float16*)allocB((size_t)Nn * INNER * 2);
  _Float16* v16t = (_Float16*)allocB((size_t)Nn * INNER * 2);
  _Float16* wt_pool = (_Float16*)allocB((size_t)900000 * 2);

  // ----- per-launch weight prep: f32 [K x Nc] -> f16 transposed [Nc x K]
  size_t woff = 0;
  auto cvtW = [&](const float* W, int K, int Nc) -> const _Float16* {
    _Float16* dst = wt_pool + woff;
    woff += (size_t)K * Nc;
    int tot = K * Nc;
    wcvt_kernel<<<(tot + 255) / 256, 256, 0, stream>>>(W, dst, K, Nc);
    return dst;
  };
  const _Float16 *iW1t[Ll], *iW2t[Ll], *uWut[Ll], *uWvt[Ll], *uWm1t[Ll], *uWm2t[Ll];
  for (int l = 0; l < Ll; ++l) {
    iW1t[l]  = cvtW(iW1  + (size_t)l * 128 * 128, 128, 128);
    iW2t[l]  = cvtW(iW2  + (size_t)l * 128 * 384, 128, 384);
    uWut[l]  = cvtW(uWu  + (size_t)l * 128 * 128, 128, 128);
    uWvt[l]  = cvtW(uWv  + (size_t)l * 128 * 128, 128, 128);
    uWm1t[l] = cvtW(uWm1 + (size_t)l * 256 * 128, 256, 128);
    uWm2t[l] = cvtW(uWm2 + (size_t)l * 128 * 384, 128, 384);
  }
  const _Float16 *Wqkvt[Tt], *fW1t[Tt], *fW2t[Tt], *Wot[Tt];
  for (int t = 0; t < Tt; ++t) {
    Wqkvt[t] = cvtW(Wqkv + (size_t)t * 128 * 384, 128, 384);
    fW1t[t]  = cvtW(fW1  + (size_t)t * 128 * 128, 128, 128);
    fW2t[t]  = cvtW(fW2  + (size_t)t * 128 * 128, 128, 128);
    Wot[t]   = cvtW(Wo   + (size_t)t * 128 * 128, 128, 128);
  }
  const _Float16* rW1t = cvtW(rW1, 128, 128);

  const int EW = Nn * Hh / 256;            // 8192 elementwise blocks
  embed_kernel<<<EW, 256, 0, stream>>>(emb, node_z, s, vA);
  edge_geom_kernel<<<Ee / 256, 256, 0, stream>>>(xyz, edges, unit, rbf, fcb);

  // ----- PaiNN interaction + update layers -----
  for (int l = 0; l < Ll; ++l) {
    gemm_wmma_kernel<1, 128><<<Nn / 128, 256, 0, stream>>>(
        s, iW1t[l], ib1 + (size_t)l * 128, bufA, Nn, 128);
    gemm_wmma_kernel<0, 128><<<Nn / 128, 256, 0, stream>>>(
        bufA, iW2t[l], ib2 + (size_t)l * 384, b384, Nn, 384);
    message_kernel<<<Nn / NPB, 128, 0, stream>>>(
        b384, vA, edges, unit, rbf, fcb,
        iWf + (size_t)l * Kk * 384, ibf + (size_t)l * 384, s, vB);
    gemm_wmma_kernel<0, 128><<<3 * Nn / 128, 256, 0, stream>>>(
        vB, uWut[l], nullptr, U, 3 * Nn, 128);
    gemm_wmma_kernel<0, 128><<<3 * Nn / 128, 256, 0, stream>>>(
        vB, uWvt[l], nullptr, V, 3 * Nn, 128);
    pack_kernel<<<EW, 256, 0, stream>>>(s, U, V, cat, bufB);
    gemm_wmma_kernel<1, 256><<<Nn / 128, 256, 0, stream>>>(
        cat, uWm1t[l], ubm1 + (size_t)l * 128, bufA, Nn, 128);
    gemm_wmma_kernel<0, 128><<<Nn / 128, 256, 0, stream>>>(
        bufA, uWm2t[l], ubm2 + (size_t)l * 384, b384, Nn, 384);
    combine_kernel<<<EW, 256, 0, stream>>>(b384, bufB, U, vB, s, vA);
  }

  // ----- Transformer layers -----
  for (int t = 0; t < Tt; ++t) {
    ln3_kernel<<<Nn / 8, 256, 0, stream>>>(
        s, ln_w, ln_b,
        a_ln_w + (size_t)t * 128, a_ln_b + (size_t)t * 128,
        f_ln_w + (size_t)t * 128, f_ln_b + (size_t)t * 128, xl, hb, fxb);
    gemm_wmma_kernel<0, 128><<<Nn / 128, 256, 0, stream>>>(
        hb, Wqkvt[t], nullptr, b384, Nn, 384);
    knvn_kernel<<<Nn * HEADS / 8, 256, 0, stream>>>(
        b384, k_ln_w + (size_t)t * 32, k_ln_b + (size_t)t * 32,
        v_ln_w + (size_t)t * 32, v_ln_b + (size_t)t * 32, q16, k16, v16t);
    gemm_wmma_kernel<2, 128><<<Nn / 128, 256, 0, stream>>>(
        fxb, fW1t[t], fb1 + (size_t)t * 128, bufA, Nn, 128);
    gemm_wmma_kernel<0, 128><<<Nn / 128, 256, 0, stream>>>(
        bufA, fW2t[t], fb2 + (size_t)t * 128, bufB, Nn, 128);
    attn_wmma_kernel<<<Nn / NAa * HEADS, 128, 0, stream>>>(q16, k16, v16t, hb);
    gemm_wmma_kernel<0, 128><<<Nn / 128, 256, 0, stream>>>(
        hb, Wot[t], bo + (size_t)t * 128, fxb, Nn, 128);
    residual_kernel<<<EW, 256, 0, stream>>>(xl, fxb, bufB, s);
  }

  // ----- readout -----
  ln1_kernel<<<Nn / 8, 256, 0, stream>>>(s, ln_w, ln_b, xl);
  gemm_wmma_kernel<1, 128><<<Nn / 128, 256, 0, stream>>>(xl, rW1t, rb1, bufA, Nn, 128);
  rowdot_kernel<<<Nn / 8, 256, 0, stream>>>(bufA, rW2, rb2, (float*)d_out);
}
